// BertSelfAttention_61014305407279
// MI455X (gfx1250) — compile-verified, matrix operand order checked by reading, not run on hardware
//
#include <hip/hip_runtime.h>
#include <hip/hip_bf16.h>
#include <math.h>

typedef __attribute__((ext_vector_type(16))) _Float16 v16h;
typedef __attribute__((ext_vector_type(8)))  float    v8f;
typedef int vs4i __attribute__((vector_size(16)));

#define SEQ   2048
#define HID   1024
#define NH    16
#define DH    64
#define BATCH 2
#define MTOT  (BATCH * SEQ)   // 4096 rows for QKV gemm

#if __has_builtin(__builtin_amdgcn_global_load_async_to_lds_b128)
#define HAVE_ASYNC_LDS 1
#endif

// ---------- fragment helpers ----------
// 16-bit A-matrix 16x32 layout: lane%16 = row, lane/16 selects K-offset (0 or 8);
// halves {koff..koff+7} then {koff+16..koff+23} -> two 16B loads.
__device__ __forceinline__ v16h load_frag16(const _Float16* p) {
    union { v16h v; uint4 u[2]; } u;
    u.u[0] = *reinterpret_cast<const uint4*>(p);
    u.u[1] = *reinterpret_cast<const uint4*>(p + 16);
    return u.v;
}

__device__ __forceinline__ v8f wmma_f16(v16h a, v16h b, v8f c) {
    return __builtin_amdgcn_wmma_f32_16x16x32_f16(
        /*neg_a=*/false, a, /*neg_b=*/false, b,
        /*c_mod=*/(short)0, c, /*reuse_a=*/false, /*reuse_b=*/false);
}

// 16-byte global -> LDS copy; async (ASYNCcnt-tracked, no VGPR round trip) when
// the gfx1250 builtin is available, else VGPR-staged fallback.
__device__ __forceinline__ void cp16_g2l(const _Float16* g, _Float16* l) {
#if defined(HAVE_ASYNC_LDS)
    __builtin_amdgcn_global_load_async_to_lds_b128(
        (__attribute__((address_space(1))) vs4i*)(unsigned long long)(const void*)g,
        (__attribute__((address_space(3))) vs4i*)(unsigned int)(unsigned long long)(void*)l,
        0, 0);
#else
    *reinterpret_cast<uint4*>(l) = *reinterpret_cast<const uint4*>(g);
#endif
}

__device__ __forceinline__ void wait_async_copies() {
#if defined(HAVE_ASYNC_LDS)
#if __has_builtin(__builtin_amdgcn_s_wait_asynccnt)
    __builtin_amdgcn_s_wait_asynccnt(0);
#else
    asm volatile("s_wait_asynccnt 0x0" ::: "memory");
#endif
#endif
}

// ---------- kernel 1: fp32 -> f16 convert + weight transpose ----------
__global__ void cvt_kernel(const float* __restrict__ X,
                           const float* __restrict__ Wq,
                           const float* __restrict__ Wk,
                           const float* __restrict__ Wv,
                           _Float16* __restrict__ Xh,
                           _Float16* __restrict__ Wtq,
                           _Float16* __restrict__ Wtk,
                           _Float16* __restrict__ Wtv) {
    const int NX = MTOT * HID;          // 4194304
    const int NW = HID * HID;           // 1048576
    int i = blockIdx.x * blockDim.x + threadIdx.x;
    int total = NX + 3 * NW;
    for (; i < total; i += gridDim.x * blockDim.x) {
        if (i < NX) {
            Xh[i] = (_Float16)X[i];
        } else {
            int j = i - NX;
            int w = j / NW;
            int r = j - w * NW;
            int n = r >> 10;
            int k = r & 1023;
            const float* W = (w == 0) ? Wq : (w == 1) ? Wk : Wv;
            _Float16* D    = (w == 0) ? Wtq : (w == 1) ? Wtk : Wtv;
            D[n * HID + k] = (_Float16)W[k * HID + n];
        }
    }
}

// ---------- kernel 2: QKV projection GEMM ----------
// one wave computes a 32(M) x 64(N) tile;  grid = (128 Mtiles, 16 Ntiles, 3 matrices)
// B-fragments reused across both M halves: 8 WMMAs per 12 b128 loads per k-step.
__global__ void qkv_gemm_kernel(const _Float16* __restrict__ Xh,
                                const _Float16* __restrict__ Wtq,
                                const _Float16* __restrict__ Wtk,
                                const _Float16* __restrict__ Wtv,
                                const float* __restrict__ bq,
                                const float* __restrict__ bk,
                                const float* __restrict__ bv,
                                _Float16* __restrict__ Qh,
                                _Float16* __restrict__ Kh,
                                _Float16* __restrict__ Vt) {
    const int z    = blockIdx.z;                 // 0=Q 1=K 2=V
    const _Float16* Wt = (z == 0) ? Wtq : (z == 1) ? Wtk : Wtv;
    const float*  bias = (z == 0) ? bq  : (z == 1) ? bk  : bv;

    const int lane = threadIdx.x & 31;
    const int nl   = lane & 15;
    const int hi   = lane >> 4;
    const int koff = hi * 8;

    const int m0 = blockIdx.x * 32;
    const int n0 = blockIdx.y * 64;

    v8f acc[2][4];
#pragma unroll
    for (int g = 0; g < 2; ++g)
#pragma unroll
        for (int f = 0; f < 4; ++f)
#pragma unroll
            for (int r = 0; r < 8; ++r) acc[g][f][r] = 0.0f;

    const _Float16* arow0 = Xh + (size_t)(m0 + nl) * HID;
    const _Float16* arow1 = Xh + (size_t)(m0 + 16 + nl) * HID;
    for (int k = 0; k < HID; k += 32) {
        v16h a0 = load_frag16(arow0 + k + koff);
        v16h a1 = load_frag16(arow1 + k + koff);
#pragma unroll
        for (int f = 0; f < 4; ++f) {
            v16h b = load_frag16(Wt + (size_t)(n0 + f * 16 + nl) * HID + k + koff);
            acc[0][f] = wmma_f16(a0, b, acc[0][f]);
            acc[1][f] = wmma_f16(a1, b, acc[1][f]);
        }
    }

    // epilogue: bias + scatter to per-head layout
#pragma unroll
    for (int f = 0; f < 4; ++f) {
        int n = n0 + f * 16 + nl;
        float bval = bias[n];
        int hh = n >> 6;
        int d  = n & 63;
#pragma unroll
        for (int g = 0; g < 2; ++g) {
#pragma unroll
            for (int r = 0; r < 8; ++r) {
                int m = m0 + g * 16 + r + hi * 8;
                int b = m >> 11;
                int s = m & 2047;
                float val = acc[g][f][r] + bval;
                int bh = b * NH + hh;
                if (z == 0) {
                    Qh[((size_t)bh * SEQ + s) * DH + d] = (_Float16)val;
                } else if (z == 1) {
                    Kh[((size_t)bh * SEQ + s) * DH + d] = (_Float16)val;
                } else {
                    Vt[((size_t)bh * DH + d) * SEQ + s] = (_Float16)val;
                }
            }
        }
    }
}

// ---------- kernel 3: flash attention ----------
// block = 8 waves, all on one (b,h); each wave owns a 16-row q-tile.
// K/V tiles (64 cols) are cooperatively staged in LDS (double-buffered) via
// async global->LDS copies, so each tile hits L2 once per block (8x cut) and
// no wave stalls on a load-wait. Row-sums come from a P x ones WMMA.
#define KBLK 64
#define NW_ATT 8

__global__ void attn_kernel(const _Float16* __restrict__ Qh,
                            const _Float16* __restrict__ Kh,
                            const _Float16* __restrict__ Vt,
                            const float* __restrict__ mask,
                            float* __restrict__ out) {
    __shared__ __align__(16) _Float16 Klds[2][KBLK * DH];      // 2 x 8 KB
    __shared__ __align__(16) _Float16 Vlds[2][DH * KBLK];      // 2 x 8 KB
    __shared__ __align__(16) _Float16 plds[NW_ATT][16 * KBLK]; // 8 x 2 KB

    const int tid  = threadIdx.x;
    const int wave = tid >> 5;
    const int lane = tid & 31;
    const int nl   = lane & 15;
    const int hi   = lane >> 4;
    const int koff = hi * 8;

    const int bh = blockIdx.y;
    const int b  = bh >> 4;
    const int h  = bh & 15;
    const int q0 = (blockIdx.x * NW_ATT + wave) * 16;

    const _Float16* Qbase = Qh + ((size_t)bh * SEQ + q0) * DH;
    const _Float16* Kbase = Kh + (size_t)bh * SEQ * DH;
    const _Float16* Vbase = Vt + (size_t)bh * DH * SEQ;
    const float*    mrow  = mask + (size_t)b * SEQ;

    const float LOG2E  = 1.4426950408889634f;
    const float KSCALE = 0.125f * LOG2E;      // 1/sqrt(64) folded into log2 domain

    // cooperative stage of one K/V tile pair into LDS buffer `buf`
    auto stage = [&](int buf, int k0) {
#pragma unroll
        for (int j = 0; j < 2; ++j) {
            int c   = tid * 2 + j;        // 0..511
            int row = c >> 3;             // 0..63
            int col = (c & 7) * 8;        // halves
            cp16_g2l(Kbase + (size_t)(k0 + row) * DH + col,
                     &Klds[buf][row * DH + col]);
            cp16_g2l(Vbase + (size_t)row * SEQ + k0 + col,
                     &Vlds[buf][row * KBLK + col]);
        }
    };

    // Q a-fragments (16 rows x 64 K), reused across the whole loop
    v16h qa[2];
#pragma unroll
    for (int j = 0; j < 2; ++j)
        qa[j] = load_frag16(Qbase + (size_t)nl * DH + j * 32 + koff);

    v16h ones;
#pragma unroll
    for (int i = 0; i < 16; ++i) ones[i] = (_Float16)1.0f;

    v8f ctx[4];
#pragma unroll
    for (int f = 0; f < 4; ++f)
#pragma unroll
        for (int r = 0; r < 8; ++r) ctx[f][r] = 0.0f;

    float m_run[8], l_run[8];
#pragma unroll
    for (int r = 0; r < 8; ++r) { m_run[r] = -1.0e30f; l_run[r] = 0.0f; }

    stage(0, 0);

    for (int it = 0; it < SEQ / KBLK; ++it) {
        const int k0  = it * KBLK;
        const int cur = it & 1;
        wait_async_copies();                  // drain this wave's async stores to LDS
        __syncthreads();                      // tile `cur` ready; everyone past `cur^1`
        if (it + 1 < SEQ / KBLK) stage(cur ^ 1, k0 + KBLK);

        const _Float16* Kt = &Klds[cur][0];
        const _Float16* Vv = &Vlds[cur][0];

        // ---- scores: 4 column-frags x 2 k-chunks = 8 WMMAs ----
        v8f S[4];
#pragma unroll
        for (int c = 0; c < 4; ++c) {
#pragma unroll
            for (int r = 0; r < 8; ++r) S[c][r] = 0.0f;
#pragma unroll
            for (int j = 0; j < 2; ++j) {
                v16h kb = load_frag16(Kt + (size_t)(16 * c + nl) * DH + j * 32 + koff);
                S[c] = wmma_f16(qa[j], kb, S[c]);
            }
        }

        float mk[4];
#pragma unroll
        for (int c = 0; c < 4; ++c) mk[c] = mrow[k0 + 16 * c + nl] * LOG2E;

        // ---- online softmax (log2 domain) ----
        float rscale[8], e[4][8];
#pragma unroll
        for (int r = 0; r < 8; ++r) {
            float v0 = S[0][r] * KSCALE + mk[0];
            float v1 = S[1][r] * KSCALE + mk[1];
            float v2 = S[2][r] * KSCALE + mk[2];
            float v3 = S[3][r] * KSCALE + mk[3];
            float mx = fmaxf(fmaxf(v0, v1), fmaxf(v2, v3));
#pragma unroll
            for (int off = 1; off < 16; off <<= 1)
                mx = fmaxf(mx, __shfl_xor(mx, off, 16));
            float mnew = fmaxf(m_run[r], mx);
            rscale[r] = exp2f(m_run[r] - mnew);
            e[0][r] = exp2f(v0 - mnew);
            e[1][r] = exp2f(v1 - mnew);
            e[2][r] = exp2f(v2 - mnew);
            e[3][r] = exp2f(v3 - mnew);
            m_run[r] = mnew;
        }

        // ---- re-layout P (C-format -> A-format) through per-wave LDS ----
        _Float16* pw = &plds[wave][0];
#pragma unroll
        for (int r = 0; r < 8; ++r) {
            int row = r + hi * 8;
#pragma unroll
            for (int c = 0; c < 4; ++c)
                pw[row * KBLK + 16 * c + nl] = (_Float16)e[c][r];
        }
        v16h pa[2];
#pragma unroll
        for (int j = 0; j < 2; ++j)
            pa[j] = load_frag16(pw + (size_t)nl * KBLK + j * 32 + koff);

        // ---- row sums via P x ones (moves reduction onto the matrix pipe) ----
        v8f ls;
#pragma unroll
        for (int r = 0; r < 8; ++r) ls[r] = 0.0f;
        ls = wmma_f16(pa[0], ones, ls);
        ls = wmma_f16(pa[1], ones, ls);

        // ---- rescale ctx, update l, ctx += P x V (8 WMMAs) ----
#pragma unroll
        for (int f = 0; f < 4; ++f)
#pragma unroll
            for (int r = 0; r < 8; ++r) ctx[f][r] *= rscale[r];
#pragma unroll
        for (int r = 0; r < 8; ++r) l_run[r] = l_run[r] * rscale[r] + ls[r];

#pragma unroll
        for (int f = 0; f < 4; ++f) {
#pragma unroll
            for (int j = 0; j < 2; ++j) {
                v16h vb = load_frag16(Vv + (size_t)(f * 16 + nl) * KBLK + j * 32 + koff);
                ctx[f] = wmma_f16(pa[j], vb, ctx[f]);
            }
        }
    }

    // ---- normalize and write [B,S,h*dh] fp32 ----
#pragma unroll
    for (int f = 0; f < 4; ++f) {
#pragma unroll
        for (int r = 0; r < 8; ++r) {
            int row = r + hi * 8;
            out[((size_t)b * SEQ + q0 + row) * HID + h * DH + f * 16 + nl] =
                ctx[f][r] / l_run[r];
        }
    }
}

// ---------- launcher ----------
extern "C" void kernel_launch(void* const* d_in, const int* in_sizes, int n_in,
                              void* d_out, int out_size, void* d_ws, size_t ws_size,
                              hipStream_t stream) {
    const float* hidden = (const float*)d_in[0];
    const float* amask  = (const float*)d_in[1];
    const float* Wq     = (const float*)d_in[2];
    const float* bq     = (const float*)d_in[3];
    const float* Wk     = (const float*)d_in[4];
    const float* bk     = (const float*)d_in[5];
    const float* Wv     = (const float*)d_in[6];
    const float* bv     = (const float*)d_in[7];
    float* out = (float*)d_out;

    // workspace layout (bytes); total ~38 MB
    char* ws = (char*)d_ws;
    const size_t XH_OFF  = 0;
    const size_t WTQ_OFF = XH_OFF  + (size_t)MTOT * HID * 2;
    const size_t WTK_OFF = WTQ_OFF + (size_t)HID * HID * 2;
    const size_t WTV_OFF = WTK_OFF + (size_t)HID * HID * 2;
    const size_t QH_OFF  = WTV_OFF + (size_t)HID * HID * 2;
    const size_t KH_OFF  = QH_OFF  + (size_t)MTOT * HID * 2;
    const size_t VT_OFF  = KH_OFF  + (size_t)MTOT * HID * 2;

    _Float16* Xh  = (_Float16*)(ws + XH_OFF);
    _Float16* Wtq = (_Float16*)(ws + WTQ_OFF);
    _Float16* Wtk = (_Float16*)(ws + WTK_OFF);
    _Float16* Wtv = (_Float16*)(ws + WTV_OFF);
    _Float16* Qh  = (_Float16*)(ws + QH_OFF);
    _Float16* Kh  = (_Float16*)(ws + KH_OFF);
    _Float16* Vt  = (_Float16*)(ws + VT_OFF);

    // 1) convert inputs to f16 (+ transpose weights)
    {
        int total  = MTOT * HID + 3 * HID * HID;
        int blocks = (total + 255) / 256;
        cvt_kernel<<<blocks, 256, 0, stream>>>(hidden, Wq, Wk, Wv, Xh, Wtq, Wtk, Wtv);
    }

    // 2) QKV projections via WMMA (32x64 tiles per wave)
    {
        dim3 grid(MTOT / 32, HID / 64, 3);   // (128, 16, 3)
        qkv_gemm_kernel<<<grid, 32, 0, stream>>>(Xh, Wtq, Wtk, Wtv, bq, bk, bv,
                                                 Qh, Kh, Vt);
    }

    // 3) flash attention via WMMA with async-LDS-staged K/V
    {
        dim3 grid(SEQ / (16 * NW_ATT), BATCH * NH);   // (16, 32)
        attn_kernel<<<grid, 256, 0, stream>>>(Qh, Kh, Vt, amask, out);
    }
}
